// MyModel_61933428410278
// MI455X (gfx1250) — matrix operand order repaired and checked by reading, not compiled, not run on hardware
//
#include <hip/hip_runtime.h>
#include <hip/hip_bf16.h>

typedef __attribute__((ext_vector_type(8)))  float    v8f;
typedef __attribute__((ext_vector_type(2)))  _Float16 h2;
typedef __attribute__((ext_vector_type(4)))  _Float16 v4h;
typedef __attribute__((ext_vector_type(8)))  _Float16 v8h;
typedef __attribute__((ext_vector_type(16))) _Float16 v16h;
typedef __attribute__((ext_vector_type(8)))  unsigned v8u;

#define B_      2
#define H_      16
#define SQ_     2048
#define SKV_    3072
#define D_      64
#define NUMPT_  1024
// 1/sqrt(64) * log2(e): softmax computed in log2 domain -> bare v_exp_f32
#define SCALE2_ (0.125f * 1.44269504088896340736f)

#define ROWS_PER_WG 128         // 8 waves * 16 q-rows
#define KV_TILE     64
#define KSTRH       72          // f16 LDS row stride (halves): 144B rows, 16B-aligned
#define ESTR        68          // f32 epilogue stride
// LDS: f16 K + f16 V^T tiles = 18432 B; f32 epilogue = 34816 B (overlaid)
#define SMEM_BYTES  34816

#if __has_builtin(__builtin_amdgcn_exp2f)
#define EXP2(x) __builtin_amdgcn_exp2f(x)
#else
#define EXP2(x) exp2f(x)
#endif

// cross-half lane swap (lane i <-> lane i^16), pure VALU
__device__ __forceinline__ float permx16(float v) {
    int iv = (int)__float_as_uint(v);
    int r  = __builtin_amdgcn_permlanex16(iv, iv, 0x76543210, (int)0xfedcba98,
                                          false, false);
    return __uint_as_float((unsigned)r);
}
__device__ __forceinline__ unsigned permx16u(unsigned v) {
    return (unsigned)__builtin_amdgcn_permlanex16((int)v, (int)v, 0x76543210,
                                                  (int)0xfedcba98, false, false);
}
// pack two f32 -> one dword of two f16 (v_cvt_pk_f16_f32)
__device__ __forceinline__ unsigned pkh2(float a, float b) {
    h2 p; p[0] = (_Float16)a; p[1] = (_Float16)b;
    return __builtin_bit_cast(unsigned, p);
}
// A/B f16 operand: 16 halves = two contiguous 8-half (16B) LDS chunks
__device__ __forceinline__ v16h ld_op16(const _Float16* base) {
    v8h lo = *(const v8h*)(base);
    v8h hi = *(const v8h*)(base + 8);
    return __builtin_shufflevector(lo, hi, 0,1,2,3,4,5,6,7,8,9,10,11,12,13,14,15);
}
// balanced-tree horizontal reductions (log-depth, ILP-friendly)
__device__ __forceinline__ float hsum8(v8f v) {
    float a = v[0] + v[1], b = v[2] + v[3], c = v[4] + v[5], d = v[6] + v[7];
    return (a + b) + (c + d);
}
__device__ __forceinline__ float hmax8(v8f v) {
    float a = fmaxf(v[0], v[1]), b = fmaxf(v[2], v[3]);
    float c = fmaxf(v[4], v[5]), d = fmaxf(v[6], v[7]);
    return fmaxf(fmaxf(a, b), fmaxf(c, d));
}

__global__ __launch_bounds__(256)
void fa_prefix_causal_f16wmma(const float* __restrict__ Q,
                              const float* __restrict__ K,
                              const float* __restrict__ V,
                              float* __restrict__ O) {
    extern __shared__ float smem[];
    _Float16* Kh = (_Float16*)smem;            // [kv][d]  f16, stride KSTRH
    _Float16* Vt = Kh + KV_TILE * KSTRH;       // [d][kv]  f16 transposed, stride KSTRH

    const int tid  = threadIdx.x;
    const int wave = tid >> 5;
    const int lane = tid & 31;
    const int hi   = lane >> 4;                // WMMA K-split lane half
    const int l15  = lane & 15;

    const int h      = blockIdx.y;
    const int b      = blockIdx.z;
    const int qblock = blockIdx.x * ROWS_PER_WG;
    const int q0w    = qblock + wave * 16;

    const size_t bh = (size_t)(b * H_ + h);
    const float* Qb = Q + bh * SQ_  * D_;
    const float* Kb = K + bh * SKV_ * D_;
    const float* Vb = V + bh * SKV_ * D_;
    float*       Ob = O + bh * SQ_  * D_;

    // ---- Q as f16 B operand of S^T = K*Q^T (log2e folded into scale) ----
    v16h bq[2];
    {
        const float* qrow = Qb + (size_t)(q0w + l15) * D_;
        #pragma unroll
        for (int si = 0; si < 2; ++si)
            #pragma unroll
            for (int j = 0; j < 16; ++j)
                bq[si][j] = (_Float16)(qrow[si * 32 + 16 * hi + j] * SCALE2_);
    }

    // O^T accumulators: oacc[dg][r] = O[q0w+l15][dg*16 + r + 8*hi]
    v8f oacc[4];
    #pragma unroll
    for (int dg = 0; dg < 4; ++dg)
        oacc[dg] = (v8f){0.f,0.f,0.f,0.f,0.f,0.f,0.f,0.f};
    float m = -__builtin_inff();               // per-lane stats (log2 domain)
    float l = 0.f;

    const int kv_end = min(SKV_, qblock + ROWS_PER_WG + NUMPT_);   // mult of 64

    for (int kv0 = 0; kv0 < kv_end; kv0 += KV_TILE) {
        __syncthreads();
        // ---- stage K (row-major f16) and V (transposed f16) ----
        #pragma unroll
        for (int it = 0; it < 4; ++it) {
            const int idx = tid + it * 256;
            const int row = idx >> 4;
            const int c4  = idx & 15;
            const float4 kd = *(const float4*)(Kb + (size_t)(kv0 + row) * D_ + c4 * 4);
            v4h k4 = { (_Float16)kd.x, (_Float16)kd.y, (_Float16)kd.z, (_Float16)kd.w };
            *(v4h*)(&Kh[row * KSTRH + c4 * 4]) = k4;
            const float4 vd = *(const float4*)(Vb + (size_t)(kv0 + row) * D_ + c4 * 4);
            Vt[(c4 * 4 + 0) * KSTRH + row] = (_Float16)vd.x;
            Vt[(c4 * 4 + 1) * KSTRH + row] = (_Float16)vd.y;
            Vt[(c4 * 4 + 2) * KSTRH + row] = (_Float16)vd.z;
            Vt[(c4 * 4 + 3) * KSTRH + row] = (_Float16)vd.w;
            if (kv0 + KV_TILE < kv_end) {
                __builtin_prefetch(Kb + (size_t)(kv0 + KV_TILE + row) * D_ + c4 * 4, 0, 0);
                __builtin_prefetch(Vb + (size_t)(kv0 + KV_TILE + row) * D_ + c4 * 4, 0, 0);
            }
        }
        __syncthreads();

        if (kv0 > q0w + 15 + NUMPT_) continue;   // whole tile masked for this wave

        // ---- S^T subtiles: 2 x f16 WMMA (K=32) each ----
        v8f s4[4];
        #pragma unroll
        for (int t = 0; t < 4; ++t) {
            const int c0 = kv0 + t * 16;
            if (c0 > q0w + 15 + NUMPT_) {        // fully masked subtile
                s4[t] = (v8f){-__builtin_inff(),-__builtin_inff(),-__builtin_inff(),
                              -__builtin_inff(),-__builtin_inff(),-__builtin_inff(),
                              -__builtin_inff(),-__builtin_inff()};
                continue;
            }
            v8f s = (v8f){0.f,0.f,0.f,0.f,0.f,0.f,0.f,0.f};
            #pragma unroll
            for (int si = 0; si < 2; ++si) {
                v16h ak = ld_op16(&Kh[(t * 16 + l15) * KSTRH + si * 32 + 16 * hi]);
                s = __builtin_amdgcn_wmma_f32_16x16x32_f16(
                        false, ak, false, bq[si], (short)0, s, false, false);
            }
            if (c0 + 15 > q0w + NUMPT_) {        // partial mask
                const int lim = q0w + l15 + NUMPT_ - c0;
                #pragma unroll
                for (int r = 0; r < 8; ++r)
                    if (r + 8 * hi > lim) s[r] = -__builtin_inff();
            }
            s4[t] = s;
        }

        // ---- tile-level online softmax in log2 domain (tree reductions) ----
        float tmax = fmaxf(fmaxf(hmax8(s4[0]), hmax8(s4[1])),
                           fmaxf(hmax8(s4[2]), hmax8(s4[3])));
        tmax = fmaxf(tmax, permx16(tmax));
        const float mn    = fmaxf(m, tmax);
        const float alpha = EXP2(m - mn);
        m = mn;
        #pragma unroll
        for (int t = 0; t < 4; ++t)
            #pragma unroll
            for (int r = 0; r < 8; ++r)
                s4[t][r] = EXP2(s4[t][r] - mn);
        float ts = (hsum8(s4[0]) + hsum8(s4[1])) + (hsum8(s4[2]) + hsum8(s4[3]));
        ts += permx16(ts);
        l = l * alpha + ts;
        #pragma unroll
        for (int dg = 0; dg < 4; ++dg)
            #pragma unroll
            for (int r = 0; r < 8; ++r) oacc[dg][r] *= alpha;

        // ---- O^T += V^T * P^T : 2 K-slices of 32 kv, 8 f16 WMMAs ----
        #pragma unroll
        for (int si = 0; si < 2; ++si) {
            // P^T B operand: pack to f16 dwords, exchange/select whole dwords
            const v8f p0 = s4[2 * si];
            const v8f p1 = s4[2 * si + 1];
            unsigned p0d[4], p1d[4];
            #pragma unroll
            for (int j = 0; j < 4; ++j) {
                p0d[j] = pkh2(p0[2 * j], p0[2 * j + 1]);
                p1d[j] = pkh2(p1[2 * j], p1[2 * j + 1]);
            }
            v8u bpu;
            #pragma unroll
            for (int j = 0; j < 4; ++j) {
                bpu[j]     = hi ? permx16u(p1d[j]) : p0d[j];
                bpu[j + 4] = hi ? p1d[j] : permx16u(p0d[j]);
            }
            const v16h bp = __builtin_bit_cast(v16h, bpu);
            #pragma unroll
            for (int dg = 0; dg < 4; ++dg) {
                v16h av = ld_op16(&Vt[(dg * 16 + l15) * KSTRH + si * 32 + 16 * hi]);
                oacc[dg] = __builtin_amdgcn_wmma_f32_16x16x32_f16(
                               false, av, false, bp, (short)0, oacc[dg], false, false);
            }
        }
    }

    // ---- epilogue: normalize, transpose O^T -> O via LDS, coalesced store ----
    __syncthreads();
    float* Ew = smem + wave * 16 * ESTR;
    const float inv = 1.0f / l;
    __builtin_amdgcn_wave_barrier();
    #pragma unroll
    for (int dg = 0; dg < 4; ++dg)
        #pragma unroll
        for (int r = 0; r < 8; ++r)
            Ew[l15 * ESTR + dg * 16 + r + 8 * hi] = oacc[dg][r] * inv;
    __builtin_amdgcn_wave_barrier();
    #pragma unroll
    for (int it = 0; it < 8; ++it) {
        const int idx = lane + it * 32;
        const int row = idx >> 4;
        const int c4  = idx & 15;
        const float4 o4 = *(const float4*)(&Ew[row * ESTR + c4 * 4]);
        *(float4*)(Ob + (size_t)(q0w + row) * D_ + c4 * 4) = o4;
    }
}

extern "C" void kernel_launch(void* const* d_in, const int* in_sizes, int n_in,
                              void* d_out, int out_size, void* d_ws, size_t ws_size,
                              hipStream_t stream) {
    const float* q = (const float*)d_in[0];
    const float* k = (const float*)d_in[1];
    const float* v = (const float*)d_in[2];
    float* out = (float*)d_out;
    dim3 grid(SQ_ / ROWS_PER_WG, H_, B_);   // (16, 16, 2)
    fa_prefix_causal_f16wmma<<<grid, 256, SMEM_BYTES, stream>>>(q, k, v, out);
}
